// XGBEmbedding_30073361006779
// MI455X (gfx1250) — matrix operand order repaired and checked by reading, not compiled, not run on hardware
//
#include <hip/hip_runtime.h>
#include <stdint.h>

#define B_  256
#define T_  128
#define N_  256
#define E_  64
#define H_  16
#define K2_ 2032   // (T-1)*H

typedef __attribute__((ext_vector_type(16))) __bf16 v16bf;
typedef __attribute__((ext_vector_type(8)))  float  v8f;

union Frag {
  v16bf          bf;
  unsigned short u16[16];
  unsigned int   u32[8];
  uint4          u128[2];
};

// native f32 -> bf16 (v_cvt_*bf16_f32 on gfx1250)
__device__ __forceinline__ unsigned short f2bf(float f) {
  __bf16 h = (__bf16)f;
  return __builtin_bit_cast(unsigned short, h);
}

__device__ __forceinline__ v8f wmma_bf16(const Frag& a, const Frag& b, v8f c) {
  return __builtin_amdgcn_wmma_f32_16x16x32_bf16(
      /*neg_a=*/false, a.bf, /*neg_b=*/false, b.bf,
      /*c_mod=*/(short)0, c, /*reuse_a=*/false, /*reuse_b=*/false);
}

// B-fragment slot for source element (k within 32-chunk, col):
//   lane = col + 16*((k>>3)&1), elem = ((k>>4)<<3) | (k&7)   -> frag[lane*16 + elem]
__device__ __forceinline__ int bslot(int k, int col) {
  return ((((k >> 3) & 1) << 4) + col) * 16 + (((k >> 4) << 3) | (k & 7));
}
// A-fragment slot for source element (row, K within 32-chunk):
//   lane = row + 16*((K>>3)&1), elem = ((K>>4)<<3) | (K&7)
__device__ __forceinline__ int aslot(int row, int K) {
  return ((((K >> 3) & 1) << 4) + row) * 16 + (((K >> 4) << 3) | (K & 7));
}

// ---- prep: shifted embedding gather + cast to bf16, layout beT[T][B][E] ----
__global__ void gather_cast_kernel(const int* __restrict__ x,
                                   const float* __restrict__ emb,
                                   unsigned short* __restrict__ beT) {
  int idx = blockIdx.x * blockDim.x + threadIdx.x;  // over T*B*E
  int j   = idx >> 14;            // B_*E_ = 16384
  int rem = idx & 16383;
  int b   = rem >> 6;
  int e   = rem & 63;
  int row = x[b * T_ + j] + j * N_;
  beT[idx] = f2bf(emb[(size_t)row * E_ + e]);
}

// ---- fused: (be@W1 -> relu) @ W2 -> z, z@Wout -> streaming LSE-CE ----
__global__ __launch_bounds__(256)
void fused_forest_kernel(const int* __restrict__ x,
                         const float* __restrict__ linear1,
                         const float* __restrict__ bias1,
                         const float* __restrict__ linear2,
                         const float* __restrict__ bias2,
                         const float* __restrict__ linear_out,
                         const unsigned short* __restrict__ beT,
                         float* __restrict__ out) {
  // all tiles stored in WMMA fragment order: [frag][lane 0..31][elem 0..15]
  __shared__ __align__(16) unsigned short sWoutF[16 * 2 * 512 / 16 * 16]; // 16 nt x 2 chunks x 512
  __shared__ __align__(16) unsigned short sW1F[4 * 512];   // (js,chunk) x 512
  __shared__ __align__(16) unsigned short sW2F[4 * 512];   // nt x 512
  __shared__ __align__(16) unsigned short sAux[8 * 1024];  // per-wave A-frag scratch

  const int i    = blockIdx.x;          // output tree index
  const int tid  = threadIdx.x;
  const int wave = tid >> 5;
  const int lane = tid & 31;
  const int col  = lane & 15;
  const int hh   = lane >> 4;           // lane half
  unsigned short* aux = &sAux[wave * 1024];

  // Wout[i] -> LDS bf16, fragment order
  {
    const float* src = linear_out + (size_t)i * (E_ * N_);
    for (int t = 0; t < (E_ * N_) / 256; ++t) {
      int idx = tid + t * 256;          // idx = k*256 + n
      int k = idx >> 8, n = idx & 255;
      int dst = ((n >> 4) * 2 + (k >> 5)) * 512 + bslot(k & 31, n & 15);
      sWoutF[dst] = f2bf(src[idx]);
    }
  }

  v8f zacc[2][4];
  for (int mt = 0; mt < 2; ++mt)
    for (int nt = 0; nt < 4; ++nt)
      for (int r = 0; r < 8; ++r) zacc[mt][nt][r] = 0.0f;

  for (int jp = 0; jp < T_ / 2; ++jp) {
    const int j0 = 2 * jp;
    __syncthreads();
    // W1 pair: 2048 contiguous f32 -> bf16 LDS fragment order
    {
      const float* src = linear1 + ((size_t)i * T_ + j0) * (E_ * H_);
      for (int t = 0; t < 8; ++t) {
        int idx = tid + t * 256;        // js*1024 + k*16 + h
        int js = idx >> 10, rem = idx & 1023;
        int k = rem >> 4, h = rem & 15;
        int dst = (js * 2 + (k >> 5)) * 512 + bslot(k & 31, h);
        sW1F[dst] = f2bf(src[idx]);
      }
    }
    // W2 pair (zeroed when j==i -> diagonal drop), fragment order
    for (int t = 0; t < 8; ++t) {
      int idx = tid + t * 256;          // 0..2047 = js*1024 + h*64 + n
      int js = idx >> 10, rem = idx & 1023;
      int h = rem >> 6, n = rem & 63;
      int j = j0 + js;
      unsigned short v = 0;
      if (j != i) {
        int jj = j - (j > i);
        v = f2bf(linear2[((size_t)i * K2_ + (size_t)jj * H_) * E_ + rem]);
      }
      int k = js * 16 + h;              // GEMM2 K index 0..31
      sW2F[(n >> 4) * 512 + bslot(k, n & 15)] = v;
    }
    __syncthreads();

    for (int mt = 0; mt < 2; ++mt) {
      const int m0 = (wave * 2 + mt) * 16;
      // A fragments (X rows) for j0/j1, two K=32 chunks each: b128 loads
      Frag a0l, a0h, a1l, a1h;
      {
        const uint4* p0 = (const uint4*)(beT + ((size_t)j0 * B_ + (m0 + col)) * E_);
        const uint4* p1 = (const uint4*)(beT + ((size_t)(j0 + 1) * B_ + (m0 + col)) * E_);
        a0l.u128[0] = p0[hh];     a0l.u128[1] = p0[2 + hh];
        a0h.u128[0] = p0[4 + hh]; a0h.u128[1] = p0[6 + hh];
        a1l.u128[0] = p1[hh];     a1l.u128[1] = p1[2 + hh];
        a1h.u128[0] = p1[4 + hh]; a1h.u128[1] = p1[6 + hh];
      }
      // B fragments: contiguous 32B per lane -> ds_load_b128 x2 each
      Frag b0l, b0h, b1l, b1h;
      {
        const uint4* q = (const uint4*)(sW1F + lane * 16);
        b0l.u128[0] = q[0];        b0l.u128[1] = q[1];
        b0h.u128[0] = q[64];       b0h.u128[1] = q[65];     // +512 ushorts
        b1l.u128[0] = q[128];      b1l.u128[1] = q[129];
        b1h.u128[0] = q[192];      b1h.u128[1] = q[193];
      }
      v8f y0 = {0.f,0.f,0.f,0.f,0.f,0.f,0.f,0.f};
      v8f y1 = {0.f,0.f,0.f,0.f,0.f,0.f,0.f,0.f};
      y0 = wmma_bf16(a0l, b0l, y0);
      y0 = wmma_bf16(a0h, b0h, y0);
      y1 = wmma_bf16(a1l, b1l, y1);
      y1 = wmma_bf16(a1h, b1h, y1);
      // bias1 + relu, write straight into A-fragment order (K = h index)
      float bb0 = bias1[((size_t)i * T_ + j0) * H_ + col];
      float bb1 = bias1[((size_t)i * T_ + j0 + 1) * H_ + col];
      for (int r = 0; r < 8; ++r) {
        int row = r + 8 * hh;
        aux[aslot(row, col)]      = f2bf(fmaxf(y0[r] + bb0, 0.f));
        aux[aslot(row, 16 + col)] = f2bf(fmaxf(y1[r] + bb1, 0.f));
      }
      asm volatile("s_wait_dscnt 0" ::: "memory");
      // A fragment for GEMM2: contiguous reload
      Frag ay;
      {
        const uint4* q = (const uint4*)(aux + lane * 16);
        ay.u128[0] = q[0]; ay.u128[1] = q[1];
      }
      // GEMM2: z += Y @ W2pair  (N=64 -> 4 tiles)
      for (int nt = 0; nt < 4; ++nt) {
        Frag bw;
        const uint4* q = (const uint4*)(sW2F + nt * 512 + lane * 16);
        bw.u128[0] = q[0]; bw.u128[1] = q[1];
        zacc[mt][nt] = wmma_bf16(ay, bw, zacc[mt][nt]);
      }
    }
  }

  // ---- epilogue: logits = z @ Wout[i], streaming logsumexp-CE ----
  for (int mt = 0; mt < 2; ++mt) {
    const int m0 = (wave * 2 + mt) * 16;
    // z (+bias2) -> A-fragment order bf16 (two K=32 chunks)
    for (int nt = 0; nt < 4; ++nt) {
      float b2 = bias2[(size_t)i * E_ + nt * 16 + col];
      for (int r = 0; r < 8; ++r) {
        int row = r + 8 * hh;
        int K   = nt * 16 + col;
        aux[(K >> 5) * 512 + aslot(row, K & 31)] = f2bf(zacc[mt][nt][r] + b2);
      }
    }
    asm volatile("s_wait_dscnt 0" ::: "memory");
    Frag azl, azh;
    {
      const uint4* q = (const uint4*)(aux + lane * 16);
      azl.u128[0] = q[0];  azl.u128[1] = q[1];
      azh.u128[0] = q[32]; azh.u128[1] = q[33];   // +256 ushorts
    }
    float rmax[8], rsum[8], pick[8];
    int xr[8];
    for (int r = 0; r < 8; ++r) {
      rmax[r] = -1e30f; rsum[r] = 0.f; pick[r] = 0.f;
      xr[r] = x[(size_t)(m0 + r + 8 * hh) * T_ + i];
    }
    for (int nt = 0; nt < 16; ++nt) {
      Frag bwl, bwh;
      const uint4* q = (const uint4*)(sWoutF + nt * 1024 + lane * 16);
      bwl.u128[0] = q[0];  bwl.u128[1] = q[1];
      bwh.u128[0] = q[32]; bwh.u128[1] = q[33];
      v8f d = {0.f,0.f,0.f,0.f,0.f,0.f,0.f,0.f};
      d = wmma_bf16(azl, bwl, d);
      d = wmma_bf16(azh, bwh, d);
      for (int r = 0; r < 8; ++r) {
        float v  = d[r];
        float nm = fmaxf(rmax[r], v);
        rsum[r]  = rsum[r] * __expf(rmax[r] - nm) + __expf(v - nm);
        rmax[r]  = nm;
        if (((xr[r] >> 4) == nt) && ((xr[r] & 15) == col)) pick[r] = v;
      }
    }
    // combine the 16 lanes holding each row
    for (int r = 0; r < 8; ++r) {
      float m = rmax[r], s = rsum[r], p = pick[r];
      for (int mask = 1; mask < 16; mask <<= 1) {
        float om = __shfl_xor(m, mask, 32);
        float os = __shfl_xor(s, mask, 32);
        float op = __shfl_xor(p, mask, 32);
        float nm = fmaxf(m, om);
        s = s * __expf(m - nm) + os * __expf(om - nm);
        m = nm;
        p += op;
      }
      if (col == 0) {
        int row = m0 + r + 8 * hh;
        out[(size_t)row * T_ + i] = m + __logf(s) - p;
      }
    }
  }
}

extern "C" void kernel_launch(void* const* d_in, const int* in_sizes, int n_in,
                              void* d_out, int out_size, void* d_ws, size_t ws_size,
                              hipStream_t stream) {
  (void)in_sizes; (void)n_in; (void)out_size; (void)ws_size;
  const int*   x   = (const int*)d_in[0];
  const float* emb = (const float*)d_in[1];
  const float* l1  = (const float*)d_in[2];
  const float* b1  = (const float*)d_in[3];
  const float* l2  = (const float*)d_in[4];
  const float* b2  = (const float*)d_in[5];
  const float* lo  = (const float*)d_in[6];
  float* out = (float*)d_out;
  unsigned short* beT = (unsigned short*)d_ws;   // 4 MB: [T][B][E] bf16

  gather_cast_kernel<<<(T_ * B_ * E_) / 256, 256, 0, stream>>>(x, emb, beT);
  fused_forest_kernel<<<T_, 256, 0, stream>>>(x, l1, b1, l2, b2, lo, beT, out);
}
